// PromptEncoder_4191888081146
// MI455X (gfx1250) — matrix operand order repaired and checked by reading, not compiled, not run on hardware
//
#include <hip/hip_runtime.h>
#include <cstdint>

// ---------------- problem constants ----------------
#define BB     32
#define NPROP  256
#define NCLS   19
#define NPTS   4096
#define NBQ    64
#define NCK    64
#define CC     256
#define QH     768
#define VQ     8
#define NOUT2  6144              // VQ*QH
#define MROWS  2048              // B*NBQ == B*NCK
#define FEAT_ELEMS 25165824      // B*1024*QH

#define PADK   40                // halfs per LDS panel row: 32 data + 8 pad (80B, 16B-aligned, 20-bank stride)
#define ABUF   (128 * PADK)      // halfs per panel buffer (A or B): 5120 (10 KB)

typedef _Float16 half16 __attribute__((ext_vector_type(16)));
typedef _Float16 half8  __attribute__((ext_vector_type(8)));
typedef float    float8 __attribute__((ext_vector_type(8)));
typedef int      int4v  __attribute__((ext_vector_type(4)));

// ---------------- gfx1250 async global->LDS staging ----------------
#if __has_builtin(__builtin_amdgcn_global_load_async_to_lds_b128)
#define HAS_ASYNC_LDS 1
typedef __attribute__((address_space(1))) int4v as1_int4;
typedef __attribute__((address_space(3))) int4v as3_int4;
#else
#define HAS_ASYNC_LDS 0
#endif

__device__ __forceinline__ void stage_chunk16(const _Float16* __restrict__ g, _Float16* l) {
#if HAS_ASYNC_LDS
  __builtin_amdgcn_global_load_async_to_lds_b128(
      (as1_int4*)(uintptr_t)g, (as3_int4*)(uintptr_t)l, 0, 0);
#else
  *(half8*)l = *(const half8*)g;   // global_load_b128 + ds_store_b128 fallback
#endif
}

__device__ __forceinline__ void wait_stage() {
#if HAS_ASYNC_LDS
#if __has_builtin(__builtin_amdgcn_s_wait_asynccnt)
  __builtin_amdgcn_s_wait_asynccnt(0);
#else
  asm volatile("s_wait_asynccnt 0x0" ::: "memory");
#endif
#endif
}

// Stage one 128x32 A panel and one 128x32 B panel (fp16) into LDS.
// 512 16B chunks per panel side; 256 threads handle 2 A + 2 B chunks each.
__device__ __forceinline__ void stage_panels(const _Float16* __restrict__ Xg,
                                             const _Float16* __restrict__ Wg, int K,
                                             int rowBase, int colBase, int kb,
                                             _Float16* As, _Float16* Bs, int tid) {
#pragma unroll
  for (int h = 0; h < 2; ++h) {
    const int c = tid + h * 256;   // 0..511
    const int r = c >> 2;          // panel row / panel col: 0..127
    const int q = c & 3;           // 16B chunk within the 64B K-slice
    stage_chunk16(Xg + (size_t)(rowBase + r) * K + kb + q * 8, As + r * PADK + q * 8);
    stage_chunk16(Wg + (size_t)(colBase + r) * K + kb + q * 8, Bs + r * PADK + q * 8);
  }
}

// ---------------- WMMA fragment loaders (from LDS panels) ----------------
// A 16x32 f16 tile (ISA 7.12.2): lanes 0-15 = rows; halves 0-7 = K[hi*8..+7],
// halves 8-15 = K[16+hi*8..+7], hi = lane>>4. Two ds_load_b128 per lane.
__device__ __forceinline__ half16 lds_a_frag(const _Float16* As, int rowOff, int lane) {
  const int row = rowOff + (lane & 15);
  const int hi  = lane >> 4;
  const _Float16* p = As + row * PADK + hi * 8;
  half8 a0 = *(const half8*)p;
  half8 a1 = *(const half8*)(p + 16);
  return __builtin_shufflevector(a0, a1, 0,1,2,3,4,5,6,7,8,9,10,11,12,13,14,15);
}

// B 32x16 f16 tile: lanes = columns (mod 16); halves 0-7 = K[hi*8..+7], halves 8-15 = K[16+hi*8..+7].
__device__ __forceinline__ half16 lds_b_frag(const _Float16* Bs, int colOff, int lane) {
  const int col = colOff + (lane & 15);
  const int hi  = lane >> 4;
  const _Float16* p = Bs + col * PADK + hi * 8;
  half8 b0 = *(const half8*)p;
  half8 b1 = *(const half8*)(p + 16);
  return __builtin_shufflevector(b0, b1, 0,1,2,3,4,5,6,7,8,9,10,11,12,13,14,15);
}

// ---------------- unified LDS-tiled WMMA GEMM ----------------
// Block: 256 threads (8 waves) -> 128x128 output tile, double-buffered 32-deep K panels.
// Wave (mw,nw) computes 32x64; 8 v_wmma_f32_16x16x32_f16 per K-step from registers.
// mode 0: out = relu(acc+bias) -> fp16 Hout[M,N]
// mode 1: out = acc+bias -> f32 Fout scattered into [B,1024,768] (batch-interleaved)
__global__ void __launch_bounds__(256)
gemm_lds(const _Float16* __restrict__ X, const _Float16* __restrict__ WT,
         const float* __restrict__ bias, _Float16* __restrict__ Hout,
         float* __restrict__ Fout, int M, int K, int N, int mode, int branchOff) {
  __shared__ __align__(16) _Float16 sm[4 * ABUF];  // [A panel x2 bufs | B panel x2 bufs]
  const int tid  = threadIdx.x;
  const int lane = tid & 31;
  const int wave = tid >> 5;
  const int mTiles = M >> 7;
  const int bt = blockIdx.x;
  if (bt >= mTiles * (N >> 7)) return;             // wave-uniform guard
  const int mb = bt % mTiles, nb = bt / mTiles;
  const int rowBase = mb << 7, colBase = nb << 7;
  const int mw = (wave & 3) << 5;                  // 0/32/64/96 rows within tile
  const int nw = (wave >> 2) << 6;                 // 0/64 cols within tile

  const float8 zero = {0.f,0.f,0.f,0.f,0.f,0.f,0.f,0.f};
  float8 acc[2][4] = {{zero,zero,zero,zero},{zero,zero,zero,zero}};

  stage_panels(X, WT, K, rowBase, colBase, 0, sm, sm + 2 * ABUF, tid);

  for (int kb = 0; kb < K; kb += 32) {
    const int buf = (kb >> 5) & 1;
    wait_stage();
    __syncthreads();                               // all panels for `buf` visible
    if (kb + 32 < K)                               // prefetch next panels into other buffer
      stage_panels(X, WT, K, rowBase, colBase, kb + 32,
                   sm + (buf ^ 1) * ABUF, sm + 2 * ABUF + (buf ^ 1) * ABUF, tid);
    const _Float16* As = sm + buf * ABUF;
    const _Float16* Bs = sm + 2 * ABUF + buf * ABUF;
    const half16 a0 = lds_a_frag(As, mw + 0,  lane);
    const half16 a1 = lds_a_frag(As, mw + 16, lane);
    half16 bf[4];
#pragma unroll
    for (int t = 0; t < 4; ++t) bf[t] = lds_b_frag(Bs, nw + t * 16, lane);
#pragma unroll
    for (int t = 0; t < 4; ++t) {
      acc[0][t] = __builtin_amdgcn_wmma_f32_16x16x32_f16(
          false, a0, false, bf[t], (short)0, acc[0][t], false, false);
      acc[1][t] = __builtin_amdgcn_wmma_f32_16x16x32_f16(
          false, a1, false, bf[t], (short)0, acc[1][t], false, false);
    }
  }

  const int hi8 = (lane >> 4) << 3;
#pragma unroll
  for (int sub = 0; sub < 2; ++sub) {
    const int r0 = rowBase + mw + sub * 16 + hi8;
#pragma unroll
    for (int t = 0; t < 4; ++t) {
      const int col = colBase + nw + t * 16 + (lane & 15);
      const float bv = bias[col];
      if (mode == 0) {
#pragma unroll
        for (int v = 0; v < 8; ++v) {
          float val = acc[sub][t][v] + bv;
          Hout[(size_t)(r0 + v) * N + col] = (_Float16)(val > 0.f ? val : 0.f);
        }
      } else {
        const size_t base = (size_t)branchOff + (size_t)(r0 >> 6) * 393216;
#pragma unroll
        for (int v = 0; v < 8; ++v)
          Fout[base + (size_t)(r0 + v) * NOUT2 + col] = acc[sub][t][v] + bv;
      }
    }
  }
}

// ---------------- K0: weight convert + transpose ----------------
__global__ void __launch_bounds__(256)
convert_wt(const float* __restrict__ W, _Float16* __restrict__ WT, int K, int N) {
  int idx = blockIdx.x * blockDim.x + threadIdx.x;
  if (idx >= K * N) return;
  int n = idx / K, k = idx - n * K;
  WT[(size_t)n * K + k] = (_Float16)W[(size_t)k * N + n];
}

// ---------------- K1: proposal AABB + semantic mask ----------------
__global__ void __launch_bounds__(256)
prop_aabb(const float* __restrict__ logits, const float* __restrict__ corners,
          float* __restrict__ mnp, float* __restrict__ mxp,
          float* __restrict__ volp, float* __restrict__ maskp) {
  int i = blockIdx.x * blockDim.x + threadIdx.x;
  if (i >= BB * NPROP) return;
  const float* lg = logits + (size_t)i * NCLS;
  float best = lg[0]; int bi = 0;
#pragma unroll
  for (int c = 1; c < NCLS; ++c) { float v = lg[c]; if (v > best) { best = v; bi = c; } }
  maskp[i] = (bi != NCLS - 1) ? 1.0f : 0.0f;
  const float* cr = corners + (size_t)i * 24;
  float mn0 = cr[0], mn1 = cr[1], mn2 = cr[2];
  float mx0 = mn0, mx1 = mn1, mx2 = mn2;
#pragma unroll
  for (int k = 1; k < 8; ++k) {
    float a = cr[k * 3 + 0], b = cr[k * 3 + 1], c = cr[k * 3 + 2];
    mn0 = fminf(mn0, a); mx0 = fmaxf(mx0, a);
    mn1 = fminf(mn1, b); mx1 = fmaxf(mx1, b);
    mn2 = fminf(mn2, c); mx2 = fmaxf(mx2, c);
  }
  mnp[i * 3 + 0] = mn0; mnp[i * 3 + 1] = mn1; mnp[i * 3 + 2] = mn2;
  mxp[i * 3 + 0] = mx0; mxp[i * 3 + 1] = mx1; mxp[i * 3 + 2] = mx2;
  volp[i] = (mx0 - mn0) * (mx1 - mn1) * (mx2 - mn2);
}

// ---------------- K2: box IoU argmax + fp16 gather (one wave per query) ----
__global__ void __launch_bounds__(256)
box_match_gather(const float* __restrict__ box_query,
                 const float* __restrict__ mnp, const float* __restrict__ mxp,
                 const float* __restrict__ volp, const float* __restrict__ maskp,
                 const float* __restrict__ prop_features,
                 _Float16* __restrict__ Xbox) {
  const int wave = (blockIdx.x * blockDim.x + threadIdx.x) >> 5;
  const int lane = threadIdx.x & 31;
  if (wave >= BB * NBQ) return;
  const int b = wave / NBQ;
  const float* qc = box_query + (size_t)wave * 24;
  float mn0 = qc[0], mn1 = qc[1], mn2 = qc[2];
  float mx0 = mn0, mx1 = mn1, mx2 = mn2;
#pragma unroll
  for (int k = 1; k < 8; ++k) {
    float a = qc[k * 3 + 0], c1 = qc[k * 3 + 1], c2 = qc[k * 3 + 2];
    mn0 = fminf(mn0, a); mx0 = fmaxf(mx0, a);
    mn1 = fminf(mn1, c1); mx1 = fmaxf(mx1, c1);
    mn2 = fminf(mn2, c2); mx2 = fmaxf(mx2, c2);
  }
  const float vq = (mx0 - mn0) * (mx1 - mn1) * (mx2 - mn2);
  float bestV = -1.0f; int bestI = 0;
#pragma unroll
  for (int j = 0; j < NPROP / 32; ++j) {
    const int p = j * 32 + lane;
    const size_t pi = (size_t)b * NPROP + p;
    const float* pm = mnp + pi * 3;
    const float* pM = mxp + pi * 3;
    float e0 = fminf(mx0, pM[0]) - fmaxf(mn0, pm[0]);
    float e1 = fminf(mx1, pM[1]) - fmaxf(mn1, pm[1]);
    float e2 = fminf(mx2, pM[2]) - fmaxf(mn2, pm[2]);
    e0 = e0 > 0.f ? e0 : 0.f; e1 = e1 > 0.f ? e1 : 0.f; e2 = e2 > 0.f ? e2 : 0.f;
    const float inter = e0 * e1 * e2;
    float iou = inter / (vq + volp[pi] - inter + 1e-8f);
    iou *= maskp[pi];
    if (iou > bestV) { bestV = iou; bestI = p; }
  }
#pragma unroll
  for (int off = 16; off >= 1; off >>= 1) {
    float ov = __shfl_xor(bestV, off, 32);
    int   oi = __shfl_xor(bestI, off, 32);
    if (ov > bestV || (ov == bestV && oi < bestI)) { bestV = ov; bestI = oi; }
  }
  const float* pf = prop_features + ((size_t)b * NPROP + bestI) * CC;
  _Float16* xr = Xbox + (size_t)wave * CC;
#pragma unroll
  for (int k = 0; k < CC / 32; ++k) { int c = k * 32 + lane; xr[c] = (_Float16)pf[c]; }
}

// ---------------- K3: click NN + Fourier pos emb (one wave per click) ------
__global__ void __launch_bounds__(256)
click_prep(const float* __restrict__ click_query, const float* __restrict__ pc_min,
           const float* __restrict__ pc_max, const float* __restrict__ enc_xyz,
           const float* __restrict__ enc_features, const float* __restrict__ gaussB,
           _Float16* __restrict__ Xclick) {
  const int wave = (blockIdx.x * blockDim.x + threadIdx.x) >> 5;
  const int lane = threadIdx.x & 31;
  if (wave >= BB * NCK) return;
  const int b = wave / NCK;
  const float* cq = click_query + (size_t)wave * 3;
  const float x = cq[0], y = cq[1], z = cq[2];
  float bestD = 3.4e38f; int bestI = 0;
  for (int j = 0; j < NPTS / 32; ++j) {
    const int p = j * 32 + lane;
    const float* e = enc_xyz + ((size_t)b * NPTS + p) * 3;
    const float dx = x - e[0], dy = y - e[1], dz = z - e[2];
    const float d2 = dx * dx + dy * dy + dz * dz;
    if (d2 < bestD) { bestD = d2; bestI = p; }
  }
#pragma unroll
  for (int off = 16; off >= 1; off >>= 1) {
    float ov = __shfl_xor(bestD, off, 32);
    int   oi = __shfl_xor(bestI, off, 32);
    if (ov < bestD || (ov == bestD && oi < bestI)) { bestD = ov; bestI = oi; }
  }
  _Float16* xr = Xclick + (size_t)wave * (2 * CC);
  const float* f = enc_features + ((size_t)b * NPTS + bestI) * CC;
#pragma unroll
  for (int k = 0; k < CC / 32; ++k) { int c = k * 32 + lane; xr[c] = (_Float16)f[c]; }
  const float TWO_PI = 6.283185307179586f;
  const float px = TWO_PI * (x - pc_min[b * 3 + 0]) / (pc_max[b * 3 + 0] - pc_min[b * 3 + 0]);
  const float py = TWO_PI * (y - pc_min[b * 3 + 1]) / (pc_max[b * 3 + 1] - pc_min[b * 3 + 1]);
  const float pz = TWO_PI * (z - pc_min[b * 3 + 2]) / (pc_max[b * 3 + 2] - pc_min[b * 3 + 2]);
#pragma unroll
  for (int t = 0; t < 4; ++t) {
    const int j = t * 32 + lane;
    const float pr = px * gaussB[j] + py * gaussB[128 + j] + pz * gaussB[256 + j];
    xr[CC + j]       = (_Float16)sinf(pr);
    xr[CC + 128 + j] = (_Float16)cosf(pr);
  }
}

// ---------------- K6: prompt mask ----------------
__global__ void __launch_bounds__(256)
write_mask(const float* __restrict__ bq, const float* __restrict__ cq,
           float* __restrict__ outMask) {
  int i = blockIdx.x * blockDim.x + threadIdx.x;
  if (i >= BB * 1024) return;
  const int b = i >> 10, j = i & 1023;
  outMask[i] = (j < 512) ? bq[b * NBQ + (j >> 3)] : cq[b * NCK + ((j - 512) >> 3)];
}

// ---------------- launcher ----------------
extern "C" void kernel_launch(void* const* d_in, const int* in_sizes, int n_in,
                              void* d_out, int out_size, void* d_ws, size_t ws_size,
                              hipStream_t stream) {
  (void)in_sizes; (void)n_in; (void)out_size; (void)ws_size;
  const float* sem_cls    = (const float*)d_in[0];
  const float* prop_feat  = (const float*)d_in[1];
  const float* box_corn   = (const float*)d_in[2];
  const float* enc_xyz    = (const float*)d_in[3];
  const float* enc_feat   = (const float*)d_in[4];
  const float* pc_min     = (const float*)d_in[5];
  const float* pc_max     = (const float*)d_in[6];
  const float* box_query  = (const float*)d_in[7];
  const float* box_qmask  = (const float*)d_in[8];
  const float* click_q    = (const float*)d_in[9];
  const float* click_qm   = (const float*)d_in[10];
  const float* gauss_B    = (const float*)d_in[11];
  const float* box_w1     = (const float*)d_in[12];
  const float* box_b1     = (const float*)d_in[13];
  const float* box_w2     = (const float*)d_in[14];
  const float* box_b2     = (const float*)d_in[15];
  const float* click_w1   = (const float*)d_in[16];
  const float* click_b1   = (const float*)d_in[17];
  const float* click_w2   = (const float*)d_in[18];
  const float* click_b2   = (const float*)d_in[19];
  float* out = (float*)d_out;

  // scratch carve-out (256B aligned slots)
  char* ws = (char*)d_ws;
  size_t off = 0;
  auto carve = [&](size_t bytes) -> char* {
    off = (off + 255) & ~(size_t)255;
    char* p = ws + off;
    off += bytes;
    return p;
  };
  float*    mnp   = (float*)   carve((size_t)BB * NPROP * 3 * 4);
  float*    mxp   = (float*)   carve((size_t)BB * NPROP * 3 * 4);
  float*    volp  = (float*)   carve((size_t)BB * NPROP * 4);
  float*    maskp = (float*)   carve((size_t)BB * NPROP * 4);
  _Float16* Xbox  = (_Float16*)carve((size_t)MROWS * CC * 2);
  _Float16* Xclk  = (_Float16*)carve((size_t)MROWS * 2 * CC * 2);
  _Float16* Hbox  = (_Float16*)carve((size_t)MROWS * QH * 2);
  _Float16* Hclk  = (_Float16*)carve((size_t)MROWS * QH * 2);
  _Float16* WT1b  = (_Float16*)carve((size_t)QH * CC * 2);
  _Float16* WT2b  = (_Float16*)carve((size_t)NOUT2 * QH * 2);
  _Float16* WT1c  = (_Float16*)carve((size_t)QH * 2 * CC * 2);
  _Float16* WT2c  = (_Float16*)carve((size_t)NOUT2 * QH * 2);

  const int TPB = 256;
  // K0: weight conversion (transposed fp16)
  convert_wt<<<(CC * QH + TPB - 1) / TPB, TPB, 0, stream>>>(box_w1, WT1b, CC, QH);
  convert_wt<<<(QH * NOUT2 + TPB - 1) / TPB, TPB, 0, stream>>>(box_w2, WT2b, QH, NOUT2);
  convert_wt<<<(2 * CC * QH + TPB - 1) / TPB, TPB, 0, stream>>>(click_w1, WT1c, 2 * CC, QH);
  convert_wt<<<(QH * NOUT2 + TPB - 1) / TPB, TPB, 0, stream>>>(click_w2, WT2c, QH, NOUT2);

  // K1: proposal AABBs + mask
  prop_aabb<<<(BB * NPROP + TPB - 1) / TPB, TPB, 0, stream>>>(
      sem_cls, box_corn, mnp, mxp, volp, maskp);

  // K2: box match + gather (2048 waves)
  box_match_gather<<<(BB * NBQ * 32 + TPB - 1) / TPB, TPB, 0, stream>>>(
      box_query, mnp, mxp, volp, maskp, prop_feat, Xbox);

  // K3: click prep (2048 waves)
  click_prep<<<(BB * NCK * 32 + TPB - 1) / TPB, TPB, 0, stream>>>(
      click_q, pc_min, pc_max, enc_xyz, enc_feat, gauss_B, Xclk);

  // K4: layer-1 GEMMs (128x128 tiles: 16*6 = 96 blocks)
  {
    int blocks = (MROWS / 128) * (QH / 128);
    gemm_lds<<<blocks, TPB, 0, stream>>>(Xbox, WT1b, box_b1, Hbox, nullptr,
                                         MROWS, CC, QH, 0, 0);
    gemm_lds<<<blocks, TPB, 0, stream>>>(Xclk, WT1c, click_b1, Hclk, nullptr,
                                         MROWS, 2 * CC, QH, 0, 0);
  }

  // K5: layer-2 GEMMs with fused output scatter (16*48 = 768 blocks)
  {
    int blocks = (MROWS / 128) * (NOUT2 / 128);
    gemm_lds<<<blocks, TPB, 0, stream>>>(Hbox, WT2b, box_b2, nullptr, out,
                                         MROWS, QH, NOUT2, 1, 0);
    gemm_lds<<<blocks, TPB, 0, stream>>>(Hclk, WT2c, click_b2, nullptr, out,
                                         MROWS, QH, NOUT2, 1, 512 * QH);
  }

  // K6: mask region after the features
  write_mask<<<(BB * 1024 + TPB - 1) / TPB, TPB, 0, stream>>>(
      box_qmask, click_qm, out + FEAT_ELEMS);
}
